// SwinBlock_32916629357043
// MI455X (gfx1250) — matrix operand order repaired and checked by reading, not compiled, not run on hardware
//
#include <hip/hip_runtime.h>
#include <hip/hip_bf16.h>

typedef __attribute__((ext_vector_type(16))) _Float16 v16h;
typedef __attribute__((ext_vector_type(8)))  _Float16 v8h;
typedef __attribute__((ext_vector_type(8)))  float    v8f;

#define TOK   25088      // 8*56*56 tokens
#define DMODEL 512

// ---------------------------------------------------------------------------
// Weight cast + transpose: src fp32 [K][N] row-major -> dst f16 [N][K]
// ---------------------------------------------------------------------------
__global__ __launch_bounds__(256)
void castT_kernel(const float* __restrict__ src, _Float16* __restrict__ dst,
                  int K, int N)
{
    size_t gid = (size_t)blockIdx.x * 256 + threadIdx.x;
    size_t tot = (size_t)K * N;
    if (gid >= tot) return;
    int n = (int)(gid % N);
    int k = (int)(gid / N);
    dst[(size_t)n * K + k] = (_Float16)src[gid];
}

// ---------------------------------------------------------------------------
// RMSNorm (over last dim 512) + cast to f16.  One block per token.
// ---------------------------------------------------------------------------
__global__ __launch_bounds__(256)
void rmsnorm_kernel(const float* __restrict__ x, const float* __restrict__ g,
                    _Float16* __restrict__ out)
{
    __shared__ float red[256];
    const int t   = blockIdx.x;
    const int tid = threadIdx.x;
    const float* xr = x + (size_t)t * DMODEL;
    float a = xr[tid * 2 + 0];
    float b = xr[tid * 2 + 1];
    red[tid] = a * a + b * b;
    __syncthreads();
#pragma unroll
    for (int s = 128; s > 0; s >>= 1) {
        if (tid < s) red[tid] += red[tid + s];
        __syncthreads();
    }
    float n   = sqrtf(red[0] * (1.0f / 512.0f));
    float inv = g[0] / fmaxf(n, 1e-8f);
    _Float16* orow = out + (size_t)t * DMODEL;
    orow[tid * 2 + 0] = (_Float16)(a * inv);
    orow[tid * 2 + 1] = (_Float16)(b * inv);
}

// ---------------------------------------------------------------------------
// WMMA fragment loaders (CDNA5 v_wmma_f32_16x16x32_f16 layouts, wave32).
// A (16x32 f16): lane L holds row (L&15); halves 0-7 <- K = k0+(L>>4)*8 .. +7
//                halves 8-15 <- K = that +16.
// B from pre-transposed Wt[N][K]: lane L holds col (L&15);
//                halves 0-7 <- K = k0+(L>>4)*8 .. +7, halves 8-15 <- K = that +16.
// ---------------------------------------------------------------------------
__device__ __forceinline__
v16h load_a_frag(const _Float16* __restrict__ A, int ld, int row0, int k0, int lane)
{
    int r  = row0 + (lane & 15);
    int kb = k0 + ((lane >> 4) << 3);
    const _Float16* p = A + (size_t)r * ld + kb;
    v8h lo = *(const v8h*)p;
    v8h hi = *(const v8h*)(p + 16);
    v16h o;
#pragma unroll
    for (int i = 0; i < 8; ++i) { o[i] = lo[i]; o[i + 8] = hi[i]; }
    return o;
}

__device__ __forceinline__
v16h load_b_frag(const _Float16* __restrict__ Bt, int ld, int n0, int k0, int lane)
{
    int n  = n0 + (lane & 15);
    int ks = k0 + ((lane >> 4) << 3);
    const _Float16* p = Bt + (size_t)n * ld + ks;
    v8h lo = *(const v8h*)p;
    v8h hi = *(const v8h*)(p + 16);
    v16h o;
#pragma unroll
    for (int i = 0; i < 8; ++i) { o[i] = lo[i]; o[i + 8] = hi[i]; }
    return o;
}

// ---------------------------------------------------------------------------
// GEMM:  C[M,N] = A[M,K](f16) * Wt[N,K](f16)^T  (+bias) (+residual fp32)
// Block 256 threads = 8 waves in a 2(M) x 4(N) grid; block tile 128x128.
// Each wave owns a 64x32 tile = 4x2 v_wmma_f32_16x16x32_f16 fragments
// (8 WMMAs per 32-K step on 12 b128 loads -> ~21 FLOP/byte of load traffic).
// Requires M%128==0, N%128==0, K%32==0.
// ---------------------------------------------------------------------------
template<int HAS_BIAS, int HAS_RES, int OUT_HALF>
__global__ __launch_bounds__(256)
void gemm_wmma_kernel(const _Float16* __restrict__ A,
                      const _Float16* __restrict__ Bt,
                      const float* __restrict__ bias,
                      const float* __restrict__ res,
                      void* __restrict__ Cout,
                      int M, int N, int K)
{
    const int lane = threadIdx.x & 31;
    const int wave = threadIdx.x >> 5;
    const int wm   = wave & 1;          // 0..1  (M direction, 64 rows each)
    const int wn   = wave >> 1;         // 0..3  (N direction, 32 cols each)
    const int m_base = blockIdx.x * 128 + wm * 64;
    const int n_base = blockIdx.y * 128 + wn * 32;

    v8f acc[4][2];
#pragma unroll
    for (int i = 0; i < 4; ++i)
#pragma unroll
        for (int j = 0; j < 2; ++j)
#pragma unroll
            for (int e = 0; e < 8; ++e) acc[i][j][e] = 0.0f;

#pragma unroll 2
    for (int k0 = 0; k0 < K; k0 += 32) {
        v16h afrag[4], bfrag[2];
#pragma unroll
        for (int fm = 0; fm < 4; ++fm)
            afrag[fm] = load_a_frag(A, K, m_base + fm * 16, k0, lane);
#pragma unroll
        for (int fn = 0; fn < 2; ++fn)
            bfrag[fn] = load_b_frag(Bt, K, n_base + fn * 16, k0, lane);
#pragma unroll
        for (int fm = 0; fm < 4; ++fm)
#pragma unroll
            for (int fn = 0; fn < 2; ++fn)
                acc[fm][fn] = __builtin_amdgcn_wmma_f32_16x16x32_f16(
                    false, afrag[fm], false, bfrag[fn],
                    (short)0, acc[fm][fn], false, false);
    }

    const int orow = (lane >> 4) << 3;   // 0 or 8
    const int ocol = lane & 15;
#pragma unroll
    for (int fm = 0; fm < 4; ++fm)
#pragma unroll
        for (int fn = 0; fn < 2; ++fn) {
            int col = n_base + fn * 16 + ocol;
            float bv = HAS_BIAS ? bias[col] : 0.0f;
#pragma unroll
            for (int e = 0; e < 8; ++e) {
                int row = m_base + fm * 16 + orow + e;
                size_t idx = (size_t)row * N + col;
                float v = acc[fm][fn][e] + bv;
                if (HAS_RES) v += res[idx];
                if (OUT_HALF) ((_Float16*)Cout)[idx] = (_Float16)v;
                else          ((float*)Cout)[idx]    = v;
            }
        }
}

// ---------------------------------------------------------------------------
// Time-window attention (m = 4).  One thread per (window, head, query-row).
// qkv: f16 [TOK,1536] (q|k|v each 512).  O: f16 [TOK,512].
// ---------------------------------------------------------------------------
__global__ __launch_bounds__(256)
void time_attn_kernel(const _Float16* __restrict__ qkv,
                      const float* __restrict__ mat,
                      _Float16* __restrict__ O)
{
    int gid  = blockIdx.x * 256 + threadIdx.x;   // 2*56*56 * 8 * 4 = 200704
    int i    = gid & 3;
    int head = (gid >> 2) & 7;
    int widx = gid >> 5;                         // 0..6271
    int n    = widx / 3136;
    int rem  = widx % 3136;
    int h    = rem / 56, w = rem % 56;

    int tok[4];
#pragma unroll
    for (int r = 0; r < 4; ++r)
        tok[r] = ((n * 4 + r) * 56 + h) * 56 + w;

    const _Float16* qp = qkv + (size_t)tok[i] * 1536 + head * 64;
    float sim[4];
#pragma unroll
    for (int j = 0; j < 4; ++j) {
        const _Float16* kp = qkv + (size_t)tok[j] * 1536 + 512 + head * 64;
        float s = 0.0f;
        for (int d = 0; d < 64; ++d) s += (float)qp[d] * (float)kp[d];
        sim[j] = s * 0.125f + mat[i - j + 3];
    }
    float mx = fmaxf(fmaxf(sim[0], sim[1]), fmaxf(sim[2], sim[3]));
    float sum = 0.0f;
#pragma unroll
    for (int j = 0; j < 4; ++j) { sim[j] = __expf(sim[j] - mx); sum += sim[j]; }
    float inv = 1.0f / sum;

    const _Float16* vp[4];
#pragma unroll
    for (int j = 0; j < 4; ++j)
        vp[j] = qkv + (size_t)tok[j] * 1536 + 1024 + head * 64;
    _Float16* op = O + (size_t)tok[i] * 512 + head * 64;
    for (int d = 0; d < 64; ++d) {
        float acc = 0.0f;
#pragma unroll
        for (int j = 0; j < 4; ++j) acc += sim[j] * (float)vp[j][d];
        op[d] = (_Float16)(acc * inv);
    }
}

// ---------------------------------------------------------------------------
// Space-window attention (m = 49).  One block (64 threads) per (window, head).
// Q/K/V tiles staged in LDS; scores in LDS to keep VGPRs bounded.
// ---------------------------------------------------------------------------
__global__ __launch_bounds__(64)
void space_attn_kernel(const _Float16* __restrict__ qkv,
                       const float* __restrict__ mat,
                       _Float16* __restrict__ O)
{
    __shared__ _Float16 Qs[49 * 64];
    __shared__ _Float16 Ks[49 * 64];
    __shared__ _Float16 Vs[49 * 64];
    __shared__ float    Ss[49 * 50];

    int head = blockIdx.x & 7;
    int win  = blockIdx.x >> 3;          // 0..511 = 8 frames * 64 windows
    int tt   = win >> 6;
    int rem  = win & 63;
    int hh   = rem >> 3, ww = rem & 7;
    int d    = threadIdx.x;              // 0..63

    for (int r = 0; r < 49; ++r) {
        int a = r / 7, b = r % 7;
        size_t tk = ((size_t)(tt * 56 + hh * 7 + a)) * 56 + (ww * 7 + b);
        const _Float16* base = qkv + tk * 1536 + head * 64 + d;
        Qs[r * 64 + d] = base[0];
        Ks[r * 64 + d] = base[512];
        Vs[r * 64 + d] = base[1024];
    }
    __syncthreads();

    if (threadIdx.x < 49) {
        int i = threadIdx.x;
        int ai = i / 7, bi = i % 7;
        float* srow = Ss + i * 50;
        float mx = -1e30f;
        for (int j = 0; j < 49; ++j) {
            float s = 0.0f;
            for (int dd = 0; dd < 64; ++dd)
                s += (float)Qs[i * 64 + dd] * (float)Ks[j * 64 + dd];
            int aj = j / 7, bj = j % 7;
            s = s * 0.125f + mat[(ai - aj + 6) * 13 + (bi - bj + 6)];
            srow[j] = s;
            mx = fmaxf(mx, s);
        }
        float sum = 0.0f;
        for (int j = 0; j < 49; ++j) {
            float e = __expf(srow[j] - mx);
            srow[j] = e;
            sum += e;
        }
        float inv = 1.0f / sum;
        size_t tk = ((size_t)(tt * 56 + hh * 7 + ai)) * 56 + (ww * 7 + bi);
        _Float16* op = O + tk * 512 + head * 64;
        for (int dd = 0; dd < 64; ++dd) {
            float acc = 0.0f;
            for (int j = 0; j < 49; ++j) acc += srow[j] * (float)Vs[j * 64 + dd];
            op[dd] = (_Float16)(acc * inv);
        }
    }
}

// ---------------------------------------------------------------------------
// GLU: g[t,c] = h[t,c] * gelu_exact(h[t,2048+c]),  h f16 [TOK,4096]
// ---------------------------------------------------------------------------
__global__ __launch_bounds__(256)
void glu_kernel(const _Float16* __restrict__ h, _Float16* __restrict__ g)
{
    size_t idx = (size_t)blockIdx.x * 256 + threadIdx.x;  // TOK*2048 exact
    size_t t = idx >> 11;
    int    c = (int)(idx & 2047);
    float a  = (float)h[t * 4096 + c];
    float xg = (float)h[t * 4096 + 2048 + c];
    float ge = xg * 0.5f * (1.0f + erff(xg * 0.70710678118f));
    g[idx] = (_Float16)(a * ge);
}

// ---------------------------------------------------------------------------
// Cyclic roll: out[t,h,w,:] = in[(t+st)%8, (h+sh)%56, (w+sw)%56, :]
// ---------------------------------------------------------------------------
__global__ __launch_bounds__(128)
void roll_kernel(const float4* __restrict__ in, float4* __restrict__ out,
                 int st, int sh, int sw)
{
    int tokid = blockIdx.x;
    int c     = threadIdx.x;             // 128 float4 = 512 floats
    int t   = tokid / 3136;
    int rem = tokid % 3136;
    int h = rem / 56, w = rem % 56;
    int ts = (t + st) & 7;
    int hs = h + sh; if (hs >= 56) hs -= 56;
    int ws = w + sw; if (ws >= 56) ws -= 56;
    size_t src = ((size_t)(ts * 56 + hs) * 56 + ws) * 128 + c;
    size_t dst = (size_t)tokid * 128 + c;
    out[dst] = in[src];
}

// ---------------------------------------------------------------------------
// Host orchestration
// ---------------------------------------------------------------------------
extern "C" void kernel_launch(void* const* d_in, const int* in_sizes, int n_in,
                              void* d_out, int out_size, void* d_ws, size_t ws_size,
                              hipStream_t stream)
{
    const float* X       = (const float*)d_in[0];
    const float* TA1_G   = (const float*)d_in[1];
    const float* TA1_QKV = (const float*)d_in[2];
    const float* TA1_OW  = (const float*)d_in[3];
    const float* TA1_OB  = (const float*)d_in[4];
    const float* TA1_MAT = (const float*)d_in[5];
    const float* TA2_G   = (const float*)d_in[6];
    const float* TA2_QKV = (const float*)d_in[7];
    const float* TA2_OW  = (const float*)d_in[8];
    const float* TA2_OB  = (const float*)d_in[9];
    const float* TA2_MAT = (const float*)d_in[10];
    const float* SA1_G   = (const float*)d_in[11];
    const float* SA1_QKV = (const float*)d_in[12];
    const float* SA1_OW  = (const float*)d_in[13];
    const float* SA1_OB  = (const float*)d_in[14];
    const float* SA1_MAT = (const float*)d_in[15];
    const float* SA2_G   = (const float*)d_in[16];
    const float* SA2_QKV = (const float*)d_in[17];
    const float* SA2_OW  = (const float*)d_in[18];
    const float* SA2_OB  = (const float*)d_in[19];
    const float* SA2_MAT = (const float*)d_in[20];
    const float* FF1_G   = (const float*)d_in[21];
    const float* FF1_W1  = (const float*)d_in[22];
    const float* FF1_B1  = (const float*)d_in[23];
    const float* FF1_W2  = (const float*)d_in[24];
    const float* FF1_B2  = (const float*)d_in[25];
    const float* FF2_G   = (const float*)d_in[26];
    const float* FF2_W1  = (const float*)d_in[27];
    const float* FF2_B1  = (const float*)d_in[28];
    const float* FF2_W2  = (const float*)d_in[29];
    const float* FF2_B2  = (const float*)d_in[30];
    (void)in_sizes; (void)n_in; (void)out_size; (void)ws_size;

    char* ws = (char*)d_ws;
    size_t off = 0;
    auto carve = [&](size_t nbytes) -> char* {
        char* p = ws + off;
        off += (nbytes + 255) & ~(size_t)255;
        return p;
    };
    _Float16* XN  = (_Float16*)carve((size_t)TOK * 512 * 2);   // rmsnorm output
    _Float16* QH  = (_Float16*)carve((size_t)TOK * 4096 * 2);  // qkv(1536) / ff hidden(4096)
    _Float16* OG  = (_Float16*)carve((size_t)TOK * 2048 * 2);  // attn-o(512) / glu(2048)
    float*    Ab  = (float*)   carve((size_t)TOK * 512 * 4);   // activation A
    float*    Bb  = (float*)   carve((size_t)TOK * 512 * 4);   // activation B
    _Float16* wt_qkv[4];
    for (int i = 0; i < 4; ++i) wt_qkv[i] = (_Float16*)carve((size_t)512 * 1536 * 2);
    _Float16* wt_ow[4];
    for (int i = 0; i < 4; ++i) wt_ow[i]  = (_Float16*)carve((size_t)512 * 512 * 2);
    _Float16* wt_w1[2];
    for (int i = 0; i < 2; ++i) wt_w1[i]  = (_Float16*)carve((size_t)512 * 4096 * 2);
    _Float16* wt_w2[2];
    for (int i = 0; i < 2; ++i) wt_w2[i]  = (_Float16*)carve((size_t)2048 * 512 * 2);

    auto castT = [&](const float* s, _Float16* dvec, int K, int N) {
        size_t tot = (size_t)K * N;
        castT_kernel<<<dim3((unsigned)((tot + 255) / 256)), 256, 0, stream>>>(s, dvec, K, N);
    };
    castT(TA1_QKV, wt_qkv[0], 512, 1536);
    castT(TA2_QKV, wt_qkv[1], 512, 1536);
    castT(SA1_QKV, wt_qkv[2], 512, 1536);
    castT(SA2_QKV, wt_qkv[3], 512, 1536);
    castT(TA1_OW,  wt_ow[0],  512, 512);
    castT(TA2_OW,  wt_ow[1],  512, 512);
    castT(SA1_OW,  wt_ow[2],  512, 512);
    castT(SA2_OW,  wt_ow[3],  512, 512);
    castT(FF1_W1,  wt_w1[0],  512, 4096);
    castT(FF2_W1,  wt_w1[1],  512, 4096);
    castT(FF1_W2,  wt_w2[0],  2048, 512);
    castT(FF2_W2,  wt_w2[1],  2048, 512);

    hipMemcpyAsync(Ab, X, (size_t)TOK * 512 * 4, hipMemcpyDeviceToDevice, stream);

    auto attn = [&](const float* src, const float* res, float* dst,
                    const float* g, _Float16* wqkvT, _Float16* wowT,
                    const float* ob, const float* mat, bool is_time) {
        rmsnorm_kernel<<<TOK, 256, 0, stream>>>(src, g, XN);
        gemm_wmma_kernel<0, 0, 1><<<dim3(196, 12), 256, 0, stream>>>(
            XN, wqkvT, nullptr, nullptr, (void*)QH, TOK, 1536, 512);
        if (is_time)
            time_attn_kernel<<<784, 256, 0, stream>>>(QH, mat, OG);
        else
            space_attn_kernel<<<4096, 64, 0, stream>>>(QH, mat, OG);
        gemm_wmma_kernel<1, 1, 0><<<dim3(196, 4), 256, 0, stream>>>(
            OG, wowT, ob, res, (void*)dst, TOK, 512, 512);
    };

    auto ff = [&](const float* src, const float* res, float* dst,
                  const float* g, _Float16* w1T, const float* b1,
                  _Float16* w2T, const float* b2) {
        rmsnorm_kernel<<<TOK, 256, 0, stream>>>(src, g, XN);
        gemm_wmma_kernel<1, 0, 1><<<dim3(196, 32), 256, 0, stream>>>(
            XN, w1T, b1, nullptr, (void*)QH, TOK, 4096, 512);
        glu_kernel<<<dim3(200704), 256, 0, stream>>>(QH, OG);
        gemm_wmma_kernel<1, 1, 0><<<dim3(196, 4), 256, 0, stream>>>(
            OG, w2T, b2, res, (void*)dst, TOK, 512, 2048);
    };

    // Block 1 (in place on A)
    attn(Ab, Ab, Ab, TA1_G, wt_qkv[0], wt_ow[0], TA1_OB, TA1_MAT, true);
    attn(Ab, Ab, Ab, SA1_G, wt_qkv[2], wt_ow[2], SA1_OB, SA1_MAT, false);
    ff  (Ab, Ab, Ab, FF1_G, wt_w1[0], FF1_B1, wt_w2[0], FF1_B2);

    // roll by shifts (-2,-3,-3): out[i] = in[(i+2)%8,(j+3)%56,(k+3)%56]
    roll_kernel<<<TOK, 128, 0, stream>>>((const float4*)Ab, (float4*)Bb, 2, 3, 3);

    // Block 2: time attn in place on B; B becomes residual x_
    attn(Bb, Bb, Bb, TA2_G, wt_qkv[1], wt_ow[1], TA2_OB, TA2_MAT, true);
    // space attn: src B, residual B (=x_), write A (preserve B)
    attn(Bb, Bb, Ab, SA2_G, wt_qkv[3], wt_ow[3], SA2_OB, SA2_MAT, false);
    // ff2: src A (post-spatial), residual B (=x_), write A
    ff  (Ab, Bb, Ab, FF2_G, wt_w1[1], FF2_B1, wt_w2[1], FF2_B2);

    // roll by shifts (2,4,4): out[i] = in[(i-2)%8,(j-4)%56,(k-4)%56]
    roll_kernel<<<TOK, 128, 0, stream>>>((const float4*)Ab, (float4*)d_out, 6, 52, 52);
}
